// NestedAttention_80058190397464
// MI455X (gfx1250) — matrix operand order repaired and checked by reading, not compiled, not run on hardware
//
#include <hip/hip_runtime.h>
#include <math.h>

// ---------------------------------------------------------------------------
// NestedAttention for MI455X (gfx1250): bf16 WMMA + async global->LDS staging.
// B=16, C=512, W=H=32, N=1024.
// ---------------------------------------------------------------------------

typedef __attribute__((ext_vector_type(16))) __bf16 v16bf;
typedef __attribute__((ext_vector_type(8)))  float  v8f;
typedef int v4i __attribute__((vector_size(16)));   // matches async builtin param

#define AS1 __attribute__((address_space(1)))
#define AS3 __attribute__((address_space(3)))

#define BATCH 16
#define CH    512
#define NSP   1024            // W*H
#define CN    (CH * NSP)      // 524288  per-batch [C,N] elems
#define NN    (NSP * NSP)     // 1048576 per-batch [N,N] elems
#define LDSROW 40             // padded LDS row: 32 bf16 data + 8 pad (80 B)

#if __has_builtin(__builtin_amdgcn_global_load_async_to_lds_b128) && \
    __has_builtin(__builtin_amdgcn_s_wait_asynccnt)
#define ASYNC_COPY 1
#else
#define ASYNC_COPY 0
#endif

union ABFrag { v16bf v; uint4 q[2]; };

__device__ __forceinline__ unsigned short f2bf(float f) {
  unsigned int u = __float_as_uint(f);
  unsigned int r = u + 0x7FFFu + ((u >> 16) & 1u);   // round-to-nearest-even
  return (unsigned short)(r >> 16);
}

// A fragment: matrix stored row-major [M, K] bf16, 16x32 tile at (row, kk).
// lanes 0-15 : M=lane,    K = kk+{0..7, 16..23}
// lanes 16-31: M=lane-16, K = kk+{8..15, 24..31}
__device__ __forceinline__ v16bf load_a_frag(const unsigned short* base, int ld,
                                             int row, int kk, int lane) {
  const unsigned short* p = base + (size_t)row * ld + kk + ((lane >= 16) ? 8 : 0);
  ABFrag f;
  f.q[0] = *reinterpret_cast<const uint4*>(p);
  f.q[1] = *reinterpret_cast<const uint4*>(p + 16);
  return f.v;
}

// B fragment from an LDS tile stored as [Ntile rows][LDSROW u16] (row = n).
// lanes 0-15 : N=lane,    K = {0..7, 16..23} ; lanes 16-31: N=lane-16, K = {8..15, 24..31}
__device__ __forceinline__ v16bf lds_b_frag(const unsigned short* buf, int j, int lane) {
  const unsigned short* p = buf + (j * 16 + (lane & 15)) * LDSROW + ((lane >= 16) ? 8 : 0);
  ABFrag f;
  f.q[0] = *reinterpret_cast<const uint4*>(p);
  f.q[1] = *reinterpret_cast<const uint4*>(p + 16);
  return f.v;
}

// 16-byte global -> LDS copy (async on gfx1250 when available).
__device__ __forceinline__ void async_copy16(const unsigned short* g, unsigned short* l) {
#if ASYNC_COPY
  __builtin_amdgcn_global_load_async_to_lds_b128(
      (AS1 v4i*)(uintptr_t)g,
      (AS3 v4i*)(unsigned int)(uintptr_t)l, 0, 0);
#else
  *reinterpret_cast<uint4*>(l) = *reinterpret_cast<const uint4*>(g);
#endif
}

__device__ __forceinline__ void stage_wait_barrier() {
#if ASYNC_COPY
  __builtin_amdgcn_s_wait_asynccnt(0);
#endif
  __syncthreads();
}

// ---------------------------------------------------------------------------
// SE gate: pooled -> fc1 -> gelu(exact) -> fc2 -> sigmoid; store gate^2.
// ---------------------------------------------------------------------------
__global__ __launch_bounds__(256) void se_gate_kernel(const float* __restrict__ x,
                               const float* __restrict__ Wfc1,
                               const float* __restrict__ Wfc2,
                               float* __restrict__ w2) {
  __shared__ float s_pool[CH];
  __shared__ float s_hid[32];
  const int b = blockIdx.x, t = threadIdx.x;
  const float* xb = x + (size_t)b * CN;
  for (int c = t; c < CH; c += 256) {
    const float4* p = reinterpret_cast<const float4*>(xb + (size_t)c * NSP);
    float s = 0.f;
    for (int i = 0; i < NSP / 4; ++i) { float4 v = p[i]; s += v.x + v.y + v.z + v.w; }
    s_pool[c] = s * (1.0f / (float)NSP);
  }
  __syncthreads();
  if (t < 32) {
    float s = 0.f;
    for (int c = 0; c < CH; ++c) s += s_pool[c] * Wfc1[t * CH + c];
    s_hid[t] = 0.5f * s * (1.0f + erff(s * 0.70710678118654752f));  // exact GELU
  }
  __syncthreads();
  for (int c = t; c < CH; c += 256) {
    float s = 0.f;
#pragma unroll
    for (int j = 0; j < 32; ++j) s += s_hid[j] * Wfc2[c * 32 + j];
    float w = 1.0f / (1.0f + expf(-s));
    w2[b * CH + c] = w * w;
  }
}

// ---------------------------------------------------------------------------
// Packing kernels (fp32 -> bf16, with layout changes)
// ---------------------------------------------------------------------------
__global__ void pack_x_kernel(const float* __restrict__ x,
                              unsigned short* __restrict__ xT) {
  int idx = blockIdx.x * 256 + threadIdx.x;           // [B,C,N] flat
  int n = idx & (NSP - 1);
  int c = (idx >> 10) & (CH - 1);
  int b = idx >> 19;
  xT[((size_t)b << 19) + ((size_t)n << 9) + c] = f2bf(x[idx]);  // -> [B,N,C]
}

__global__ void pack_w_kernel(const float* __restrict__ Wq,
                              const float* __restrict__ Wk,
                              const float* __restrict__ Wv,
                              unsigned short* __restrict__ W3) {
  int idx = blockIdx.x * 256 + threadIdx.x;           // 3*512*512
  int which = idx / (CH * CH);
  int r = idx - which * (CH * CH);
  const float* src = (which == 0) ? Wq : (which == 1) ? Wk : Wv;
  W3[idx] = f2bf(src[r]);
}

__global__ void pack_pos_kernel(const float* __restrict__ rel_h,
                                const float* __restrict__ rel_w,
                                unsigned short* __restrict__ posT) {
  int idx = blockIdx.x * 256 + threadIdx.x;           // [N,C] flat
  int c = idx & (CH - 1);
  int n = idx >> 9;
  int w = n >> 5, h = n & 31;
  posT[idx] = f2bf(rel_h[c * 32 + h] + rel_w[c * 32 + w]);
}

// ---------------------------------------------------------------------------
// QKV GEMM: out[o,n] = sum_c W[o,c] * x[b,c,n] + bias[o]
//   A = W3[which] [512,512] row-major (global, reg double-buffered);
//   B tile (64 n-rows x 32 k) staged in LDS, double-buffered, async copy.
//   q -> qT [B,N,C]; k -> kT [B,N,C]; v -> natural [B,C,N]
// ---------------------------------------------------------------------------
__global__ __launch_bounds__(256) void qkv_gemm_kernel(
    const unsigned short* __restrict__ W3,
    const unsigned short* __restrict__ xT,
    const float* __restrict__ bq, const float* __restrict__ bk,
    const float* __restrict__ bv,
    unsigned short* __restrict__ qT, unsigned short* __restrict__ kT,
    unsigned short* __restrict__ vn) {
  __shared__ unsigned short sB[2][64 * LDSROW];
  const int t = threadIdx.x;
  const int lane = t & 31, wave = t >> 5;
  const int lane15 = lane & 15, hi8 = (lane >= 16) ? 8 : 0;
  const int b = blockIdx.z / 3, which = blockIdx.z % 3;
  const int M0 = blockIdx.y * 128 + wave * 16;
  const int N0 = blockIdx.x * 64;
  const unsigned short* A  = W3 + (size_t)which * CH * CH;
  const unsigned short* Bt = xT + (size_t)b * CN;

  // cooperative staging coords: 64 rows x 64B, one 16B chunk per thread
  const int srow = t >> 2, scg = t & 3;
  const unsigned short* gB = Bt + (size_t)(N0 + srow) * CH + scg * 8;
  unsigned short* lB[2] = { &sB[0][srow * LDSROW + scg * 8],
                            &sB[1][srow * LDSROW + scg * 8] };

  v8f zero = {0.f, 0.f, 0.f, 0.f, 0.f, 0.f, 0.f, 0.f};
  v8f acc[4] = {zero, zero, zero, zero};

  async_copy16(gB, lB[0]);
  v16bf a_cur = load_a_frag(A, CH, M0 + lane15, 0, lane);
  stage_wait_barrier();

  for (int k0 = 0; k0 < CH; k0 += 32) {
    const int cur = (k0 >> 5) & 1;
    const bool more = (k0 + 32) < CH;
    if (more) async_copy16(gB + k0 + 32, lB[1 - cur]);       // prefetch next tile
    v16bf a_next = a_cur;
    if (more) a_next = load_a_frag(A, CH, M0 + lane15, k0 + 32, lane);
    // preload all B fragments, then issue WMMAs (partial dscnt waits)
    v16bf b0 = lds_b_frag(&sB[cur][0], 0, lane);
    v16bf b1 = lds_b_frag(&sB[cur][0], 1, lane);
    v16bf b2 = lds_b_frag(&sB[cur][0], 2, lane);
    v16bf b3 = lds_b_frag(&sB[cur][0], 3, lane);
    acc[0] = __builtin_amdgcn_wmma_f32_16x16x32_bf16(false, a_cur, false, b0, (short)0, acc[0], false, false);
    acc[1] = __builtin_amdgcn_wmma_f32_16x16x32_bf16(false, a_cur, false, b1, (short)0, acc[1], false, false);
    acc[2] = __builtin_amdgcn_wmma_f32_16x16x32_bf16(false, a_cur, false, b2, (short)0, acc[2], false, false);
    acc[3] = __builtin_amdgcn_wmma_f32_16x16x32_bf16(false, a_cur, false, b3, (short)0, acc[3], false, false);
    a_cur = a_next;
    stage_wait_barrier();
  }

  const float* bias = (which == 0) ? bq : (which == 1) ? bk : bv;
#pragma unroll
  for (int j = 0; j < 4; ++j) {
    const int n = N0 + j * 16 + lane15;
    if (which == 2) {
#pragma unroll
      for (int r = 0; r < 8; ++r) {
        const int o = M0 + r + hi8;
        vn[(size_t)b * CN + (size_t)o * NSP + n] = f2bf(acc[j][r] + bias[o]);
      }
    } else {
      unsigned short* dst = (which == 0) ? qT : kT;
#pragma unroll
      for (int r = 0; r < 8; r += 2) {   // o, o+1 contiguous -> pack dword
        const int o = M0 + r + hi8;
        unsigned int lo = f2bf(acc[j][r]     + bias[o]);
        unsigned int hi = f2bf(acc[j][r + 1] + bias[o + 1]);
        *reinterpret_cast<unsigned int*>(
            &dst[(size_t)b * CN + (size_t)n * CH + o]) = lo | (hi << 16);
      }
    }
  }
}

// ---------------------------------------------------------------------------
// Fused logits GEMM: L[n,m] = sum_{c'=0}^{1023} [q;pos][c',n] * [k;q][c',m]
//   A rows (n) from qT (k<512) / posT (k>=512); B rows (m) from kT / qT.
//   B tile (128 m-rows x 32 k) staged in LDS (double-buffered, async).
// ---------------------------------------------------------------------------
__global__ __launch_bounds__(256) void logits_gemm_kernel(
    const unsigned short* __restrict__ qT,
    const unsigned short* __restrict__ kT,
    const unsigned short* __restrict__ posT,
    float* __restrict__ logits) {
  __shared__ unsigned short sB[2][128 * LDSROW];
  const int t = threadIdx.x;
  const int lane = t & 31, wave = t >> 5;
  const int lane15 = lane & 15, hi8 = (lane >= 16) ? 8 : 0;
  const int b = blockIdx.z;
  const int M0 = blockIdx.y * 128 + wave * 16;   // n rows
  const int N0 = blockIdx.x * 128;               // m cols
  const unsigned short* qTb = qT + (size_t)b * CN;
  const unsigned short* kTb = kT + (size_t)b * CN;

  const int srow = t >> 2, scg = t & 3;          // rows srow, srow+64

  v8f zero = {0.f, 0.f, 0.f, 0.f, 0.f, 0.f, 0.f, 0.f};
  v8f acc[8] = {zero, zero, zero, zero, zero, zero, zero, zero};

  // prologue: stage k=0 tile from kT
  {
    const unsigned short* g0 = kTb + (size_t)(N0 + srow) * CH + scg * 8;
    async_copy16(g0,                   &sB[0][srow * LDSROW + scg * 8]);
    async_copy16(g0 + (size_t)64 * CH, &sB[0][(srow + 64) * LDSROW + scg * 8]);
  }
  v16bf a_cur = load_a_frag(qTb, CH, M0 + lane15, 0, lane);
  stage_wait_barrier();

  for (int k0 = 0; k0 < 2 * CH; k0 += 32) {
    const int cur = (k0 >> 5) & 1;
    const int k1 = k0 + 32;
    const bool more = k1 < 2 * CH;
    if (more) {
      const unsigned short* Bn = (k1 < CH) ? kTb : qTb;
      const int kk1 = k1 & (CH - 1);
      const unsigned short* g = Bn + (size_t)(N0 + srow) * CH + kk1 + scg * 8;
      async_copy16(g,                   &sB[1 - cur][srow * LDSROW + scg * 8]);
      async_copy16(g + (size_t)64 * CH, &sB[1 - cur][(srow + 64) * LDSROW + scg * 8]);
    }
    v16bf a_next = a_cur;
    if (more) {
      const unsigned short* An = (k1 < CH) ? qTb : posT;
      a_next = load_a_frag(An, CH, M0 + lane15, k1 & (CH - 1), lane);
    }
    // two groups of 4: preload fragments, then WMMAs
#pragma unroll
    for (int jj = 0; jj < 8; jj += 4) {
      v16bf b0 = lds_b_frag(&sB[cur][0], jj + 0, lane);
      v16bf b1 = lds_b_frag(&sB[cur][0], jj + 1, lane);
      v16bf b2 = lds_b_frag(&sB[cur][0], jj + 2, lane);
      v16bf b3 = lds_b_frag(&sB[cur][0], jj + 3, lane);
      acc[jj + 0] = __builtin_amdgcn_wmma_f32_16x16x32_bf16(false, a_cur, false, b0, (short)0, acc[jj + 0], false, false);
      acc[jj + 1] = __builtin_amdgcn_wmma_f32_16x16x32_bf16(false, a_cur, false, b1, (short)0, acc[jj + 1], false, false);
      acc[jj + 2] = __builtin_amdgcn_wmma_f32_16x16x32_bf16(false, a_cur, false, b2, (short)0, acc[jj + 2], false, false);
      acc[jj + 3] = __builtin_amdgcn_wmma_f32_16x16x32_bf16(false, a_cur, false, b3, (short)0, acc[jj + 3], false, false);
    }
    a_cur = a_next;
    stage_wait_barrier();
  }

  float* Lg = logits + (size_t)b * NN;
#pragma unroll
  for (int j = 0; j < 8; ++j) {
    const int m = N0 + j * 16 + lane15;
#pragma unroll
    for (int r = 0; r < 8; ++r) {
      const int n = M0 + r + hi8;
      Lg[(size_t)n * NSP + m] = acc[j][r];
    }
  }
}

// ---------------------------------------------------------------------------
// Softmax over m (row length 1024): wave32 shfl reductions, bf16 out.
// ---------------------------------------------------------------------------
__global__ __launch_bounds__(256) void softmax_kernel(const float* __restrict__ logits,
                               unsigned short* __restrict__ att) {
  __shared__ float smax[8];
  __shared__ float ssum[8];
  const int t = threadIdx.x, lane = t & 31, wave = t >> 5;
  const size_t off = (size_t)blockIdx.x * NSP;
  float4 v = reinterpret_cast<const float4*>(logits + off)[t];

  float r = fmaxf(fmaxf(v.x, v.y), fmaxf(v.z, v.w));
#pragma unroll
  for (int m = 16; m > 0; m >>= 1) r = fmaxf(r, __shfl_xor(r, m, 32));
  if (lane == 0) smax[wave] = r;
  __syncthreads();
  float mx = smax[0];
#pragma unroll
  for (int i = 1; i < 8; ++i) mx = fmaxf(mx, smax[i]);

  float e0 = __expf(v.x - mx), e1 = __expf(v.y - mx);
  float e2 = __expf(v.z - mx), e3 = __expf(v.w - mx);
  float s = e0 + e1 + e2 + e3;
#pragma unroll
  for (int m = 16; m > 0; m >>= 1) s += __shfl_xor(s, m, 32);
  if (lane == 0) ssum[wave] = s;
  __syncthreads();
  float tot = 0.f;
#pragma unroll
  for (int i = 0; i < 8; ++i) tot += ssum[i];
  const float inv = 1.0f / tot;

  ushort4 o;
  o.x = f2bf(e0 * inv); o.y = f2bf(e1 * inv); o.z = f2bf(e2 * inv); o.w = f2bf(e3 * inv);
  reinterpret_cast<ushort4*>(att + off)[t] = o;
}

// ---------------------------------------------------------------------------
// Output GEMM: out[c,n] = (sum_m v[c,m] * attn[n,m]) * gate^2[c]
//   A = v natural [C, N(m)] row-major; B tile (64 n-rows x 32 m) in LDS.
// ---------------------------------------------------------------------------
__global__ __launch_bounds__(256) void out_gemm_kernel(
    const unsigned short* __restrict__ vn,
    const unsigned short* __restrict__ att,
    const float* __restrict__ w2,
    float* __restrict__ out) {
  __shared__ unsigned short sB[2][64 * LDSROW];
  const int t = threadIdx.x;
  const int lane = t & 31, wave = t >> 5;
  const int lane15 = lane & 15, hi8 = (lane >= 16) ? 8 : 0;
  const int b = blockIdx.z;
  const int M0 = blockIdx.y * 128 + wave * 16;   // c
  const int N0 = blockIdx.x * 64;                // n
  const unsigned short* A  = vn  + (size_t)b * CN;   // ld = 1024
  const unsigned short* Bt = att + (size_t)b * NN;   // ld = 1024

  const int srow = t >> 2, scg = t & 3;
  const unsigned short* gB = Bt + (size_t)(N0 + srow) * NSP + scg * 8;
  unsigned short* lB[2] = { &sB[0][srow * LDSROW + scg * 8],
                            &sB[1][srow * LDSROW + scg * 8] };

  v8f zero = {0.f, 0.f, 0.f, 0.f, 0.f, 0.f, 0.f, 0.f};
  v8f acc[4] = {zero, zero, zero, zero};

  async_copy16(gB, lB[0]);
  v16bf a_cur = load_a_frag(A, NSP, M0 + lane15, 0, lane);
  stage_wait_barrier();

  for (int k0 = 0; k0 < NSP; k0 += 32) {
    const int cur = (k0 >> 5) & 1;
    const bool more = (k0 + 32) < NSP;
    if (more) async_copy16(gB + k0 + 32, lB[1 - cur]);
    v16bf a_next = a_cur;
    if (more) a_next = load_a_frag(A, NSP, M0 + lane15, k0 + 32, lane);
    v16bf b0 = lds_b_frag(&sB[cur][0], 0, lane);
    v16bf b1 = lds_b_frag(&sB[cur][0], 1, lane);
    v16bf b2 = lds_b_frag(&sB[cur][0], 2, lane);
    v16bf b3 = lds_b_frag(&sB[cur][0], 3, lane);
    acc[0] = __builtin_amdgcn_wmma_f32_16x16x32_bf16(false, a_cur, false, b0, (short)0, acc[0], false, false);
    acc[1] = __builtin_amdgcn_wmma_f32_16x16x32_bf16(false, a_cur, false, b1, (short)0, acc[1], false, false);
    acc[2] = __builtin_amdgcn_wmma_f32_16x16x32_bf16(false, a_cur, false, b2, (short)0, acc[2], false, false);
    acc[3] = __builtin_amdgcn_wmma_f32_16x16x32_bf16(false, a_cur, false, b3, (short)0, acc[3], false, false);
    a_cur = a_next;
    stage_wait_barrier();
  }

  float* outb = out + (size_t)b * CN;
#pragma unroll
  for (int j = 0; j < 4; ++j) {
    const int n = N0 + j * 16 + lane15;
#pragma unroll
    for (int r = 0; r < 8; ++r) {
      const int c = M0 + r + hi8;
      outb[(size_t)c * NSP + n] = acc[j][r] * w2[b * CH + c];
    }
  }
}

// ---------------------------------------------------------------------------
// Launch
// ---------------------------------------------------------------------------
extern "C" void kernel_launch(void* const* d_in, const int* in_sizes, int n_in,
                              void* d_out, int out_size, void* d_ws, size_t ws_size,
                              hipStream_t stream) {
  const float* x     = (const float*)d_in[0];
  const float* Wq    = (const float*)d_in[1];
  const float* bq    = (const float*)d_in[2];
  const float* Wk    = (const float*)d_in[3];
  const float* bk    = (const float*)d_in[4];
  const float* Wv    = (const float*)d_in[5];
  const float* bv    = (const float*)d_in[6];
  const float* rel_h = (const float*)d_in[7];
  const float* rel_w = (const float*)d_in[8];
  const float* Wfc1  = (const float*)d_in[9];
  const float* Wfc2  = (const float*)d_in[10];
  float* out = (float*)d_out;

  // Workspace carve-up (all 16B-aligned)
  unsigned short* xT   = (unsigned short*)d_ws;         // [B,N,C]   16 MB
  unsigned short* qT   = xT + (size_t)BATCH * CN;       // [B,N,C]   16 MB
  unsigned short* kT   = qT + (size_t)BATCH * CN;       // [B,N,C]   16 MB
  unsigned short* vn   = kT + (size_t)BATCH * CN;       // [B,C,N]   16 MB
  unsigned short* W3   = vn + (size_t)BATCH * CN;       // [3,C,C]   1.5 MB
  unsigned short* posT = W3 + 3 * CH * CH;              // [N,C]     1 MB
  float* w2            = (float*)(posT + (size_t)NSP * CH);  // [B,C] 32 KB
  float* logits        = w2 + BATCH * CH;               // [B,N,N]   64 MB
  unsigned short* att  = (unsigned short*)(logits + (size_t)BATCH * NN); // 32 MB

  se_gate_kernel<<<BATCH, 256, 0, stream>>>(x, Wfc1, Wfc2, w2);
  pack_x_kernel<<<(BATCH * CN) / 256, 256, 0, stream>>>(x, xT);
  pack_w_kernel<<<(3 * CH * CH) / 256, 256, 0, stream>>>(Wq, Wk, Wv, W3);
  pack_pos_kernel<<<(NSP * CH) / 256, 256, 0, stream>>>(rel_h, rel_w, posT);

  qkv_gemm_kernel<<<dim3(NSP / 64, CH / 128, BATCH * 3), 256, 0, stream>>>(
      W3, xT, bq, bk, bv, qT, kT, vn);

  logits_gemm_kernel<<<dim3(NSP / 128, NSP / 128, BATCH), 256, 0, stream>>>(
      qT, kT, posT, logits);

  softmax_kernel<<<BATCH * NSP, 256, 0, stream>>>(logits, att);

  out_gemm_kernel<<<dim3(NSP / 64, CH / 128, BATCH), 256, 0, stream>>>(
      vn, att, w2, out);
}
